// OctreeResBlock_60043642798685
// MI455X (gfx1250) — matrix operand order repaired and checked by reading, not compiled, not run on hardware
//
#include <hip/hip_runtime.h>
#include <hip/hip_bf16.h>

typedef __attribute__((ext_vector_type(16))) __bf16 v16bf;
typedef __attribute__((ext_vector_type(8)))  __bf16 v8bf;
typedef __attribute__((ext_vector_type(8)))  float  v8f;
typedef __attribute__((ext_vector_type(4)))  int    v4i;

#define WMMA_BF16(a, b, c) \
  __builtin_amdgcn_wmma_f32_16x16x32_bf16(false, (a), false, (b), (short)0, (c), false, false)

// ---------------------------------------------------------------------------
// gfx1250 async global->LDS copy (per-lane gather, ASYNCcnt tracked), with a
// compile-safe fallback to a normal global load + ds store.
// Probe-discovered signature: (addrspace(1) v4i*, addrspace(3) v4i*, imm, imm).
#if __has_builtin(__builtin_amdgcn_global_load_async_to_lds_b128)
#define HAVE_ASYNC_LDS 1
typedef v4i __attribute__((address_space(1)))* async_gptr_t;
typedef v4i __attribute__((address_space(3)))* async_lptr_t;
__device__ __forceinline__ void copy16_to_lds(const __bf16* g, __bf16* l) {
  __builtin_amdgcn_global_load_async_to_lds_b128(
      (async_gptr_t)(__bf16*)g, (async_lptr_t)l, 0, 0);
}
#else
#define HAVE_ASYNC_LDS 0
__device__ __forceinline__ void copy16_to_lds(const __bf16* g, __bf16* l) {
  *(v8bf*)l = *(const v8bf*)g;
}
#endif

#if __has_builtin(__builtin_amdgcn_s_wait_asynccnt)
#define WAIT_ASYNC() __builtin_amdgcn_s_wait_asynccnt(0)
#else
#define WAIT_ASYNC() asm volatile("s_wait_asynccnt 0x0" ::: "memory")
#endif

// Build a 16-bit A-matrix 16x32 fragment for this lane from a row pointer.
// ISA 7.12.2: lanes 0-15 hold M=lane, elements K = [c0..c0+7, c0+16..c0+23]
// with c0 = 0 for lanes 0-15 and c0 = 8 for lanes 16-31 (caller adds it).
__device__ __forceinline__ v16bf make_a_frag(const __bf16* rowp, int c0) {
  v8bf lo = *(const v8bf*)(rowp + c0);
  v8bf hi = *(const v8bf*)(rowp + c0 + 16);
  v16bf a;
#pragma unroll
  for (int i = 0; i < 8; ++i) { a[i] = lo[i]; a[i + 8] = hi[i]; }
  return a;
}

// ---------------------------------------------------------------------------
__global__ void zero_kernel(float* p, int n) {
  int i = blockIdx.x * blockDim.x + threadIdx.x;
  if (i < n) p[i] = 0.0f;
}

// ---------------------------------------------------------------------------
// Repack row-major fp32 weights [batch, K, Nc] into bf16 WMMA B-fragments.
// Fragment index f = (b*KC + kc)*NT + nt ; per-lane 16 bf16:
//   lane n<16 : column nt*16+n, K = kc*32 + [0..7,16..23] ; lane n+16 : K = kc*32 + [8..15,24..31]
__global__ void pack_b_kernel(const float* __restrict__ src, __bf16* __restrict__ dst,
                              int batch, int K, int Nc) {
  int t = blockIdx.x * blockDim.x + threadIdx.x;
  int KC = K >> 5, NT = Nc >> 4;
  int total = batch * KC * NT * 32;
  if (t >= total) return;
  int lane = t & 31;
  int f = t >> 5;
  int nt = f % NT;
  int kc = (f / NT) % KC;
  int b  = f / (NT * KC);
  int n = nt * 16 + (lane & 15);
  int kbase = kc * 32 + ((lane >= 16) ? 8 : 0);
  __bf16* d = dst + (size_t)f * 512 + (size_t)lane * 16;
  const float* s = src + (size_t)b * K * Nc;
#pragma unroll
  for (int i = 0; i < 16; ++i)
    d[i] = (__bf16)s[(size_t)(kbase + i + ((i >= 8) ? 8 : 0)) * Nc + n];
}

// ---------------------------------------------------------------------------
// GEMM1: t_out[N,64] = data[N,256] @ w1. No LDS, no barriers: each lane loads
// its own fp32 A elements (clamped rows; garbage rows masked at store), each
// wave owns a 16-row strip and all 4 column tiles (4 accumulators).
__global__ __launch_bounds__(256) void gemm1_kernel(const float* __restrict__ data,
                                                    const __bf16* __restrict__ pw1,
                                                    float* __restrict__ t_out, int N) {
  int tid = threadIdx.x, wave = tid >> 5, lane = tid & 31;
  int n0 = blockIdx.x * 128 + wave * 16;
  int lrow = n0 + (lane & 15);
  if (lrow >= N) lrow = N - 1;                 // clamp: safe, rows >= N masked below
  const float* rp = data + (size_t)lrow * 256;
  int c0 = (lane >= 16) ? 8 : 0;

  v8f acc[4] = {};
#pragma unroll
  for (int kc = 0; kc < 8; ++kc) {
    float4 f0 = *(const float4*)(rp + kc * 32 + c0);
    float4 f1 = *(const float4*)(rp + kc * 32 + c0 + 4);
    float4 f2 = *(const float4*)(rp + kc * 32 + c0 + 16);
    float4 f3 = *(const float4*)(rp + kc * 32 + c0 + 20);
    v16bf a;
    a[0]  = (__bf16)f0.x; a[1]  = (__bf16)f0.y; a[2]  = (__bf16)f0.z; a[3]  = (__bf16)f0.w;
    a[4]  = (__bf16)f1.x; a[5]  = (__bf16)f1.y; a[6]  = (__bf16)f1.z; a[7]  = (__bf16)f1.w;
    a[8]  = (__bf16)f2.x; a[9]  = (__bf16)f2.y; a[10] = (__bf16)f2.z; a[11] = (__bf16)f2.w;
    a[12] = (__bf16)f3.x; a[13] = (__bf16)f3.y; a[14] = (__bf16)f3.z; a[15] = (__bf16)f3.w;
#pragma unroll
    for (int j = 0; j < 4; ++j) {
      v16bf b = *(const v16bf*)(pw1 + ((size_t)(kc * 4 + j) * 32 + lane) * 16);
      acc[j] = WMMA_BF16(a, b, acc[j]);
    }
  }
  int rbase = n0 + ((lane >= 16) ? 8 : 0);
#pragma unroll
  for (int j = 0; j < 4; ++j) {
    int col = j * 16 + (lane & 15);
#pragma unroll
    for (int r = 0; r < 8; ++r) {
      int row = rbase + r;
      if (row < N) t_out[(size_t)row * 64 + col] = acc[j][r];
    }
  }
}

// ---------------------------------------------------------------------------
// Column sum/sumsq over t[N,64] with atomics into sums/sumsq (pre-zeroed).
__global__ __launch_bounds__(256) void stats64_kernel(const float* __restrict__ t, int N,
                                                      float* __restrict__ sums,
                                                      float* __restrict__ sumsq) {
  int c = threadIdx.x & 63;
  int rsub = threadIdx.x >> 6;
  float s = 0.f, q = 0.f;
  for (int row = blockIdx.x * 4 + rsub; row < N; row += gridDim.x * 4) {
    float v = t[(size_t)row * 64 + c];
    s += v; q += v * v;
  }
  __shared__ float sh[2][256];
  sh[0][threadIdx.x] = s; sh[1][threadIdx.x] = q;
  __syncthreads();
  if (threadIdx.x < 64) {
    s = sh[0][threadIdx.x] + sh[0][threadIdx.x + 64] + sh[0][threadIdx.x + 128] + sh[0][threadIdx.x + 192];
    q = sh[1][threadIdx.x] + sh[1][threadIdx.x + 64] + sh[1][threadIdx.x + 128] + sh[1][threadIdx.x + 192];
    atomicAdd(&sums[threadIdx.x], s);
    atomicAdd(&sumsq[threadIdx.x], q);
  }
}

// BN affine folding: a = g*rsqrt(var+eps), b' = b - mean*a (biased variance).
__global__ void finalize_kernel(const float* sums, const float* sumsq,
                                const float* g, const float* b,
                                float* ab, int C, float invN) {
  int c = blockIdx.x * blockDim.x + threadIdx.x;
  if (c >= C) return;
  float m = sums[c] * invN;
  float v = sumsq[c] * invN - m * m;
  float a = g[c] * rsqrtf(v + 1e-3f);
  ab[c] = a;
  ab[C + c] = b[c] - m * a;
}

// h = bf16(gelu_exact(t*a + b')) over N*64 elements.
__global__ __launch_bounds__(256) void bn_gelu_kernel(const float* __restrict__ t,
                                                      const float* __restrict__ ab,
                                                      __bf16* __restrict__ h, long long total) {
  long long i = (long long)blockIdx.x * blockDim.x + threadIdx.x;
  if (i >= total) return;
  int c = (int)(i & 63);
  float x = t[i] * ab[c] + ab[64 + c];
  float gl = 0.5f * x * (1.0f + erff(x * 0.70710678118654752f));
  h[i] = (__bf16)gl;
}

// ---------------------------------------------------------------------------
// OctreeConv: t_out[n,:] = sum_k h[neigh[n,k],:] @ w3[k]  (bf16 WMMA, fp32 acc)
// Block = 64 nodes, 128 threads = 4 waves. Each wave owns a 2x2 tile grid
// (2 M-subtiles x 2 N-tiles -> 4 accumulators): 8 WMMAs per tap per wave from
// 4 A-fragment LDS loads. Gathered rows stream into double-buffered LDS via
// async global->LDS (when available), overlapped with WMMA on the other buffer.
__global__ __launch_bounds__(128) void conv_kernel(const __bf16* __restrict__ h,
                                                   const int* __restrict__ neigh,
                                                   const __bf16* __restrict__ pw3,
                                                   float* __restrict__ t_out, int N) {
  __shared__ int sIdx[27 * 64];
  __shared__ __align__(16) __bf16 sA[2][64 * 80];  // 64 rows, stride 80 (160B)
  int n0 = blockIdx.x * 64;
  int tid = threadIdx.x;
  int wave = tid >> 5, lane = tid & 31;
  int mpair = wave >> 1, npair = wave & 1;

  for (int i = tid; i < 27 * 64; i += 128) {
    int m = i / 27, k = i - m * 27;
    int row = n0 + m;
    sIdx[k * 64 + m] = (row < N) ? neigh[(size_t)row * 27 + k] : 0;
  }
  __syncthreads();

  auto stage = [&](int k, int bsel) {
    int r = tid >> 1, chb = (tid & 1) * 4;            // 2 threads/row, 64B each
    int src = sIdx[k * 64 + r];
    const __bf16* g = h + (size_t)src * 64 + chb * 8;
    __bf16* l = &sA[bsel][r * 80 + chb * 8];
#pragma unroll
    for (int c = 0; c < 4; ++c) copy16_to_lds(g + c * 8, l + c * 8);
  };

  v8f acc[2][2] = {};
  stage(0, 0);
  for (int k = 0; k < 27; ++k) {
    WAIT_ASYNC();            // my async writes for buffer k are in LDS
    __syncthreads();         // everyone's writes visible; prior reads of buf' done
    if (k + 1 < 27) stage(k + 1, (k + 1) & 1);
    const __bf16* base = sA[k & 1];
    int c0h = (lane >= 16) ? 8 : 0;
#pragma unroll
    for (int cc = 0; cc < 2; ++cc) {
      v16bf a0 = make_a_frag(base + ((mpair * 2 + 0) * 16 + (lane & 15)) * 80, cc * 32 + c0h);
      v16bf a1 = make_a_frag(base + ((mpair * 2 + 1) * 16 + (lane & 15)) * 80, cc * 32 + c0h);
      v16bf b0 = *(const v16bf*)(pw3 + ((size_t)((k * 2 + cc) * 4 + npair * 2 + 0) * 32 + lane) * 16);
      v16bf b1 = *(const v16bf*)(pw3 + ((size_t)((k * 2 + cc) * 4 + npair * 2 + 1) * 32 + lane) * 16);
      acc[0][0] = WMMA_BF16(a0, b0, acc[0][0]);
      acc[0][1] = WMMA_BF16(a0, b1, acc[0][1]);
      acc[1][0] = WMMA_BF16(a1, b0, acc[1][0]);
      acc[1][1] = WMMA_BF16(a1, b1, acc[1][1]);
    }
  }
#pragma unroll
  for (int mi = 0; mi < 2; ++mi) {
    int rbase = n0 + (mpair * 2 + mi) * 16 + ((lane >= 16) ? 8 : 0);
#pragma unroll
    for (int ni = 0; ni < 2; ++ni) {
      int col = (npair * 2 + ni) * 16 + (lane & 15);
#pragma unroll
      for (int r = 0; r < 8; ++r) {
        int row = rbase + r;
        if (row < N) t_out[(size_t)row * 64 + col] = acc[mi][ni][r];
      }
    }
  }
}

// ---------------------------------------------------------------------------
// GEMM2: t3 = h2[N,64] @ w2[64,256]. mode 0: column stats only. mode 1: fuse
// BN (ab3) + residual(data) + ELU and write out[N,256]. t3 never materialized.
__global__ __launch_bounds__(256) void gemm2_kernel(const __bf16* __restrict__ h2,
                                                    const __bf16* __restrict__ pw2,
                                                    const float* __restrict__ data,
                                                    const float* __restrict__ ab3,
                                                    float* __restrict__ out,
                                                    float* __restrict__ sums,
                                                    float* __restrict__ sumsq,
                                                    int N, int mode) {
  int n0 = blockIdx.x * 32;
  int tid = threadIdx.x;
  int wave = tid >> 5, lane = tid & 31;
  int msub = wave >> 2, ntq = wave & 3;

  int lrow = n0 + msub * 16 + (lane & 15);
  const __bf16* rp = h2 + (size_t)((lrow < N) ? lrow : 0) * 64;  // clamp; masked later
  int c0h = (lane >= 16) ? 8 : 0;

  v8f acc[4] = {};
#pragma unroll
  for (int kc = 0; kc < 2; ++kc) {
    v16bf a = make_a_frag(rp, kc * 32 + c0h);
#pragma unroll
    for (int j = 0; j < 4; ++j) {
      int ntl = ntq * 4 + j;
      v16bf b = *(const v16bf*)(pw2 + ((size_t)(kc * 16 + ntl) * 32 + lane) * 16);
      acc[j] = WMMA_BF16(a, b, acc[j]);
    }
  }

  int rbase = n0 + msub * 16 + ((lane >= 16) ? 8 : 0);
  if (mode == 0) {
    __shared__ float ss[256], sq[256];
    ss[tid] = 0.f; sq[tid] = 0.f;
    __syncthreads();
#pragma unroll
    for (int j = 0; j < 4; ++j) {
      int col = (ntq * 4 + j) * 16 + (lane & 15);
      float s = 0.f, q = 0.f;
#pragma unroll
      for (int r = 0; r < 8; ++r) {
        if (rbase + r < N) { float v = acc[j][r]; s += v; q += v * v; }
      }
      atomicAdd(&ss[col], s);
      atomicAdd(&sq[col], q);
    }
    __syncthreads();
    atomicAdd(&sums[tid], ss[tid]);
    atomicAdd(&sumsq[tid], sq[tid]);
  } else {
#pragma unroll
    for (int j = 0; j < 4; ++j) {
      int col = (ntq * 4 + j) * 16 + (lane & 15);
      float a = ab3[col], bb = ab3[256 + col];
#pragma unroll
      for (int r = 0; r < 8; ++r) {
        int row = rbase + r;
        if (row < N) {
          float v = acc[j][r] * a + bb + data[(size_t)row * 256 + col];
          out[(size_t)row * 256 + col] = (v > 0.f) ? v : expm1f(v);
        }
      }
    }
  }
}

// ---------------------------------------------------------------------------
extern "C" void kernel_launch(void* const* d_in, const int* in_sizes, int n_in,
                              void* d_out, int out_size, void* d_ws, size_t ws_size,
                              hipStream_t stream) {
  (void)n_in; (void)out_size; (void)ws_size;
  const float* data = (const float*)d_in[0];
  const int*   neigh = (const int*)d_in[1];
  const float* w1 = (const float*)d_in[2];
  const float* g1 = (const float*)d_in[3];
  const float* b1 = (const float*)d_in[4];
  const float* w3 = (const float*)d_in[5];
  const float* g3 = (const float*)d_in[6];
  const float* b3 = (const float*)d_in[7];
  const float* w2 = (const float*)d_in[8];
  const float* g2 = (const float*)d_in[9];
  const float* b2 = (const float*)d_in[10];
  float* out = (float*)d_out;

  const int C = 256;
  int N = in_sizes[0] / C;

  char* ws = (char*)d_ws;
  size_t off = 0;
  float*  t_buf  = (float*)(ws + off);  off += (size_t)N * 64 * 4;
  __bf16* h_buf  = (__bf16*)(ws + off); off += (size_t)N * 64 * 2;
  __bf16* h2_buf = (__bf16*)(ws + off); off += (size_t)N * 64 * 2;
  __bf16* pw1    = (__bf16*)(ws + off); off += (size_t)(8 * 4) * 512 * 2;       // 32KB
  __bf16* pw3    = (__bf16*)(ws + off); off += (size_t)(27 * 2 * 4) * 512 * 2;  // 216KB
  __bf16* pw2    = (__bf16*)(ws + off); off += (size_t)(2 * 16) * 512 * 2;      // 32KB
  float*  stats  = (float*)(ws + off);  off += 768 * 4;
  float*  ab     = (float*)(ws + off);  off += 768 * 4;
  float* sum1 = stats;       float* sq1 = stats + 64;
  float* sum2 = stats + 128; float* sq2 = stats + 192;
  float* sum3 = stats + 256; float* sq3 = stats + 512;
  float* ab1 = ab; float* ab2 = ab + 128; float* ab3 = ab + 256;

  int nb1 = (N + 127) / 128;   // gemm1: 128 rows/block
  int nbc = (N + 63) / 64;     // conv : 64 rows/block
  int nb2 = (N + 31) / 32;     // gemm2: 32 rows/block
  float invN = 1.0f / (float)N;
  long long tot = (long long)N * 64;
  int gbn = (int)((tot + 255) / 256);

  zero_kernel<<<3, 256, 0, stream>>>(stats, 768);
  pack_b_kernel<<<4, 256, 0, stream>>>(w1, pw1, 1, 256, 64);
  pack_b_kernel<<<27, 256, 0, stream>>>(w3, pw3, 27, 64, 64);
  pack_b_kernel<<<4, 256, 0, stream>>>(w2, pw2, 1, 64, 256);

  gemm1_kernel<<<nb1, 256, 0, stream>>>(data, pw1, t_buf, N);
  stats64_kernel<<<1024, 256, 0, stream>>>(t_buf, N, sum1, sq1);
  finalize_kernel<<<1, 64, 0, stream>>>(sum1, sq1, g1, b1, ab1, 64, invN);
  bn_gelu_kernel<<<gbn, 256, 0, stream>>>(t_buf, ab1, h_buf, tot);

  conv_kernel<<<nbc, 128, 0, stream>>>(h_buf, neigh, pw3, t_buf, N);
  stats64_kernel<<<1024, 256, 0, stream>>>(t_buf, N, sum2, sq2);
  finalize_kernel<<<1, 64, 0, stream>>>(sum2, sq2, g3, b3, ab2, 64, invN);
  bn_gelu_kernel<<<gbn, 256, 0, stream>>>(t_buf, ab2, h2_buf, tot);

  gemm2_kernel<<<nb2, 256, 0, stream>>>(h2_buf, pw2, data, ab3, out, sum3, sq3, N, 0);
  finalize_kernel<<<1, 256, 0, stream>>>(sum3, sq3, g2, b2, ab3, 256, invN);
  gemm2_kernel<<<nb2, 256, 0, stream>>>(h2_buf, pw2, data, ab3, out, sum3, sq3, N, 1);
}
